// MPNN_60198261621123
// MI455X (gfx1250) — compile-verified, matrix operand order checked by reading, not run on hardware
//
#include <hip/hip_runtime.h>
#include <hip/hip_bf16.h>
#include <math.h>

#define NATOM 10000
#define NPAIR 400000
#define NWAVE 8
#define NCONTRACT 64
#define NANG 9

typedef __attribute__((ext_vector_type(16))) __bf16 v16bf;
typedef __attribute__((ext_vector_type(8)))  __bf16 v8bf;
typedef __attribute__((ext_vector_type(8)))  float  v8f;

// ---------------------------------------------------------------------------
// small helpers
// ---------------------------------------------------------------------------
static __device__ __forceinline__ float siluf(float x) {
    return x / (1.0f + expf(-x));
}

// layernorm over 8 features + SiLU (for the tiny embedding MLPs, per-thread)
static __device__ __forceinline__ void ln8_silu(const float* pre, const float* g,
                                                const float* be, float* out) {
    float mu = 0.0f;
#pragma unroll
    for (int j = 0; j < 8; ++j) mu += pre[j];
    mu *= 0.125f;
    float var = 0.0f;
#pragma unroll
    for (int j = 0; j < 8; ++j) { float d = pre[j] - mu; var += d * d; }
    var *= 0.125f;
    float rinv = rsqrtf(var + 1e-5f);
#pragma unroll
    for (int j = 0; j < 8; ++j) {
        float xn = g[j] * (pre[j] - mu) * rinv + be[j];
        out[j] = siluf(xn);
    }
}

// hidden part of the [1,8,8] MLPs (scalar input -> 8 hidden features)
static __device__ __forceinline__ void mlp_1_8_8(
    float s,
    const float* W0, const float* b0, const float* g0, const float* be0,
    const float* W1, const float* b1, const float* g1, const float* be1,
    float* h1) {
    float pre[8], h0[8];
#pragma unroll
    for (int j = 0; j < 8; ++j) pre[j] = s * W0[j] + b0[j];
    ln8_silu(pre, g0, be0, h0);
#pragma unroll
    for (int j = 0; j < 8; ++j) {
        float a = b1[j];
#pragma unroll
        for (int i = 0; i < 8; ++i) a += h0[i] * W1[i * 8 + j];
        pre[j] = a;
    }
    ln8_silu(pre, g1, be1, h1);
}

// build a 16-bit WMMA fragment: lane holds K chunk [k0+8g .. +7] and [k0+16+8g .. +7]
static __device__ __forceinline__ v16bf frag_load(const __bf16* row64, int k0, int g) {
    const v8bf lo = *reinterpret_cast<const v8bf*>(row64 + k0 + 8 * g);
    const v8bf hi = *reinterpret_cast<const v8bf*>(row64 + k0 + 16 + 8 * g);
    return __builtin_shufflevector(lo, hi, 0, 1, 2, 3, 4, 5, 6, 7,
                                   8, 9, 10, 11, 12, 13, 14, 15);
}

// ---------------------------------------------------------------------------
// kernels
// ---------------------------------------------------------------------------
__global__ void k_zero(float* p, size_t n) {
    size_t i = (size_t)blockIdx.x * blockDim.x + threadIdx.x;
    if (i < n) p[i] = 0.0f;
}

// f32 (K x Nsrc) -> bf16 transposed (Npad x K), zero-padding extra N columns
__global__ void k_convert_wT(const float* __restrict__ src, __bf16* __restrict__ dst,
                             int K, int Nsrc, int Npad) {
    int i = blockIdx.x * blockDim.x + threadIdx.x;
    if (i >= K * Npad) return;
    int n = i / K, k = i % K;
    float v = (n < Nsrc) ? src[(size_t)k * Nsrc + n] : 0.0f;
    dst[(size_t)n * K + k] = (__bf16)v;
}

// center embedding MLP: species (NATOM,1) -> center_coeff (NATOM,64)
__global__ void k_center_embed(const float* __restrict__ species,
    const float* W0, const float* b0, const float* g0, const float* be0,
    const float* W1, const float* b1, const float* g1, const float* be1,
    const float* Wout, const float* bout, float* __restrict__ ccen) {
    int i = blockIdx.x * blockDim.x + threadIdx.x;
    if (i >= NATOM) return;
    float h1[8];
    mlp_1_8_8(species[i], W0, b0, g0, be0, W1, b1, g1, be1, h1);
    for (int m = 0; m < 64; ++m) {
        float a = bout[m];
#pragma unroll
        for (int q = 0; q < 8; ++q) a += h1[q] * Wout[q * 64 + m];
        ccen[(size_t)i * 64 + m] = a;
    }
}

// per-pair basis: distances, neighbor embedding MLP, cutoff, radial, sph
__global__ void k_pair_setup(const float* __restrict__ cart, const int* __restrict__ nlist,
    const float* __restrict__ shifts, const float* __restrict__ nfac,
    const float* __restrict__ species,
    const float* W0, const float* b0, const float* g0, const float* be0,
    const float* W1, const float* b1, const float* g1, const float* be1,
    const float* Wout, const float* bout,
    float* __restrict__ cutd, float* __restrict__ radial,
    float* __restrict__ sphv, float* __restrict__ wemb, float* __restrict__ icoef) {
    int p = blockIdx.x * blockDim.x + threadIdx.x;
    if (p >= NPAIR) return;
    int ic = nlist[p];
    int in = nlist[NPAIR + p];
    float dx = cart[in * 3 + 0] - cart[ic * 3 + 0] + shifts[(size_t)p * 3 + 0];
    float dy = cart[in * 3 + 1] - cart[ic * 3 + 1] + shifts[(size_t)p * 3 + 1];
    float dz = cart[in * 3 + 2] - cart[ic * 3 + 2] + shifts[(size_t)p * 3 + 2];
    float d = sqrtf(dx * dx + dy * dy + dz * dz);

    float s0 = species[ic], s1 = species[in];
    float hs = s0 * s1 / (s0 + s1);
    float h1[8];
    mlp_1_8_8(hs, W0, b0, g0, be0, W1, b1, g1, be1, h1);
    float ne[24];
    for (int o = 0; o < 24; ++o) {
        float a = bout[o];
#pragma unroll
        for (int q = 0; q < 8; ++q) a += h1[q] * Wout[q * 24 + o];
        ne[o] = a;
    }
    float t = 0.5f * cosf(d * 0.78539816339744831f) + 0.5f;  // pi/CUTOFF = pi/4
    cutd[p] = nfac[p] * t * t;
#pragma unroll
    for (int j = 0; j < 8; ++j) {
        float z = ne[8 + j] * (d - ne[16 + j]);
        radial[(size_t)p * 8 + j] = expf(-z * z);
        wemb[(size_t)p * 8 + j] = ne[j];
        icoef[(size_t)p * 8 + j] = ne[j];
    }
    float inv = 1.0f / d;
    float x = dx * inv, y = dy * inv, z = dz * inv;
    float* s = sphv + (size_t)p * 9;
    s[0] = 0.28209479177387814f;
    s[1] = 0.4886025119029199f * y;
    s[2] = 0.4886025119029199f * z;
    s[3] = 0.4886025119029199f * x;
    s[4] = 1.0925484305920792f * x * y;
    s[5] = 1.0925484305920792f * y * z;
    s[6] = 0.31539156525252005f * (3.0f * z * z - 1.0f);
    s[7] = 1.0925484305920792f * x * z;
    s[8] = 0.5462742152960396f * (x * x - y * y);
}

// iteration-0 segment_sum: wo[p,k,j] = cut*radial_j*sph_k*wemb_j -> corb[idx_c]
__global__ void k_scatter0(const int* __restrict__ nlist, const float* __restrict__ cutd,
                           const float* __restrict__ radial, const float* __restrict__ sphv,
                           const float* __restrict__ wemb, float* __restrict__ corb) {
    int p = blockIdx.x * blockDim.x + threadIdx.x;
    if (p >= NPAIR) return;
    int ic = nlist[p];
    float cd = cutd[p];
    float rj[8], sk[9];
#pragma unroll
    for (int j = 0; j < 8; ++j) rj[j] = radial[(size_t)p * 8 + j] * wemb[(size_t)p * 8 + j] * cd;
#pragma unroll
    for (int k = 0; k < 9; ++k) sk[k] = sphv[(size_t)p * 9 + k];
    float* dst = corb + (size_t)ic * 72;
#pragma unroll
    for (int k = 0; k < 9; ++k)
#pragma unroll
        for (int j = 0; j < 8; ++j) atomicAdd(&dst[k * 8 + j], sk[k] * rj[j]);
}

// message-passing segment_sum: iter_coeff update + gather of previous center_orbital
__global__ void k_scatter_iter(const int* __restrict__ nlist, const float* __restrict__ cutd,
                               const float* __restrict__ radial, const float* __restrict__ sphv,
                               const float* __restrict__ nnY /* NATOM x 16 */,
                               float* __restrict__ icoef,
                               const float* __restrict__ corbPrev, float* __restrict__ corbNext) {
    int p = blockIdx.x * blockDim.x + threadIdx.x;
    if (p >= NPAIR) return;
    int ic = nlist[p];
    int in = nlist[NPAIR + p];
    float cd = cutd[p];
    float rj[8];
#pragma unroll
    for (int j = 0; j < 8; ++j) {
        float v = icoef[(size_t)p * 8 + j] + nnY[(size_t)in * 16 + j];
        icoef[(size_t)p * 8 + j] = v;
        rj[j] = v * radial[(size_t)p * 8 + j] * cd;
    }
    const float* prev = corbPrev + (size_t)in * 72;
    float* dst = corbNext + (size_t)ic * 72;
#pragma unroll
    for (int k = 0; k < 9; ++k) {
        float s = sphv[(size_t)p * 9 + k];
#pragma unroll
        for (int j = 0; j < 8; ++j)
            atomicAdd(&dst[k * 8 + j], s * rj[j] + prev[k * 8 + j] * cd);
    }
}

// co = einsum('ikj,kjm->ikm'); density[i,m] = sum_k co^2 * ccen[i,m] (bf16 out for GEMM)
__global__ void k_co_density(const float* __restrict__ corb, const float* __restrict__ cc_all,
                             int it, const float* __restrict__ ccen, __bf16* __restrict__ densb) {
    const int i = blockIdx.x;
    const int m = threadIdx.x;  // 64 threads
    __shared__ float cs[72];
    for (int idx = m; idx < 72; idx += 64) cs[idx] = corb[(size_t)i * 72 + idx];
    __syncthreads();
    const int IL[9] = {0, 1, 1, 1, 2, 2, 2, 2, 2};
    float acc = 0.0f;
#pragma unroll
    for (int k = 0; k < 9; ++k) {
        const float* cp = cc_all + (size_t)((it * 3 + IL[k]) * 8) * 64 + m;
        float co = 0.0f;
#pragma unroll
        for (int j = 0; j < 8; ++j) co += cs[k * 8 + j] * cp[(size_t)j * 64];
        acc += co * co;
    }
    densb[(size_t)i * 64 + m] = (__bf16)(acc * ccen[(size_t)i * 64 + m]);
}

// WMMA GEMM: Y(M x NPAD, f32) = A(M x 64, bf16) @ Bt^T (Bt is NPAD x 64 bf16) + bias.
// One wave computes a full 16 x NPAD stripe: A fragments loaded once, reused across
// NPAD/16 B tiles -> 2*NPAD/16 back-to-back v_wmma_f32_16x16x32_bf16 per wave.
// Requires M % 16 == 0 (NATOM = 625 * 16).
template <int NPAD>
__global__ void k_gemm_bias_wmma(const __bf16* __restrict__ A, const __bf16* __restrict__ Bt,
                                 const float* __restrict__ bias, int bias_len,
                                 float* __restrict__ Y) {
    constexpr int NT = NPAD / 16;
    const int lane = threadIdx.x;
    const int g = lane >> 4;
    const int h = lane & 15;
    const int row = blockIdx.x * 16 + h;
    const __bf16* Ap = A + (size_t)row * 64;
    const v16bf a0 = frag_load(Ap, 0, g);
    const v16bf a1 = frag_load(Ap, 32, g);
    v8f c[NT];
#pragma unroll
    for (int t = 0; t < NT; ++t) {
        const __bf16* Bp = Bt + (size_t)(t * 16 + h) * 64;
        const v16bf b0 = frag_load(Bp, 0, g);
        const v16bf b1 = frag_load(Bp, 32, g);
        v8f acc = {};
        acc = __builtin_amdgcn_wmma_f32_16x16x32_bf16(false, a0, false, b0,
                                                      (short)0, acc, false, false);
        acc = __builtin_amdgcn_wmma_f32_16x16x32_bf16(false, a1, false, b1,
                                                      (short)0, acc, false, false);
        c[t] = acc;
    }
    const int mbase = blockIdx.x * 16 + 8 * g;
#pragma unroll
    for (int t = 0; t < NT; ++t) {
        const int n = t * 16 + h;
        const float bv = (n < bias_len) ? bias[n] : 0.0f;
#pragma unroll
        for (int r = 0; r < 8; ++r) {
            Y[(size_t)(mbase + r) * NPAD + n] = c[t][r] + bv;
        }
    }
}

// row-wise layernorm(64) + SiLU, f32 in -> bf16 out (next GEMM's A operand)
__global__ void k_ln_silu(const float* __restrict__ Y, const float* __restrict__ gamma,
                          const float* __restrict__ beta, __bf16* __restrict__ Xb) {
    const int i = blockIdx.x;
    const int t = threadIdx.x;  // 64
    float y = Y[(size_t)i * 64 + t];
    __shared__ float red[64];
    red[t] = y;
    __syncthreads();
    for (int s = 32; s > 0; s >>= 1) { if (t < s) red[t] += red[t + s]; __syncthreads(); }
    float mu = red[0] * (1.0f / 64.0f);
    __syncthreads();
    float d = y - mu;
    red[t] = d * d;
    __syncthreads();
    for (int s = 32; s > 0; s >>= 1) { if (t < s) red[t] += red[t + s]; __syncthreads(); }
    float var = red[0] * (1.0f / 64.0f);
    float xn = gamma[t] * d * rsqrtf(var + 1e-5f) + beta[t];
    Xb[(size_t)i * 64 + t] = (__bf16)siluf(xn);
}

// final scalar: sum_i outY[i,0] * center_factor[i]
__global__ void k_final(const float* __restrict__ outY, const float* __restrict__ cfac,
                        float* __restrict__ out) {
    int i = blockIdx.x * blockDim.x + threadIdx.x;
    float v = (i < NATOM) ? outY[(size_t)i * 16] * cfac[i] : 0.0f;
#pragma unroll
    for (int off = 16; off > 0; off >>= 1) v += __shfl_down(v, off, 32);
    if ((threadIdx.x & 31) == 0) atomicAdd(out, v);
}

// ---------------------------------------------------------------------------
// host driver
// ---------------------------------------------------------------------------
extern "C" void kernel_launch(void* const* d_in, const int* in_sizes, int n_in,
                              void* d_out, int out_size, void* d_ws, size_t ws_size,
                              hipStream_t stream) {
    (void)in_sizes; (void)n_in; (void)ws_size;
    const float* cart    = (const float*)d_in[0];
    const int*   nlist   = (const int*)d_in[1];
    const float* shifts  = (const float*)d_in[2];
    const float* cfac    = (const float*)d_in[3];
    const float* nfac    = (const float*)d_in[4];
    const float* species = (const float*)d_in[5];
    const float* cc_all  = (const float*)d_in[6];
    // nested-dict flattening order: W0,W1,b0,b1,g0,g1,be0,be1,Wout,bout
    const int EN = 7, EC = 17, IT0 = 27, OUTP = 57;
    auto F = [&](int i) { return (const float*)d_in[i]; };

    char* base = (char*)d_ws;
    size_t off = 0;
    auto alloc = [&](size_t bytes) -> char* {
        char* p = base + off;
        off = (off + bytes + 255) & ~(size_t)255;
        return p;
    };
    float*  cutd   = (float*)alloc((size_t)NPAIR * 4);
    float*  radial = (float*)alloc((size_t)NPAIR * 8 * 4);
    float*  sphv   = (float*)alloc((size_t)NPAIR * 9 * 4);
    float*  wemb   = (float*)alloc((size_t)NPAIR * 8 * 4);
    float*  icoef  = (float*)alloc((size_t)NPAIR * 8 * 4);
    float*  ccen   = (float*)alloc((size_t)NATOM * 64 * 4);
    float*  corbA  = (float*)alloc((size_t)NATOM * 72 * 4);
    float*  corbB  = (float*)alloc((size_t)NATOM * 72 * 4);
    __bf16* densb  = (__bf16*)alloc((size_t)NATOM * 64 * 2);
    float*  Yf     = (float*)alloc((size_t)NATOM * 64 * 4);
    __bf16* Xb     = (__bf16*)alloc((size_t)NATOM * 64 * 2);
    float*  nnY    = (float*)alloc((size_t)NATOM * 16 * 4);
    __bf16* wbf    = (__bf16*)alloc((size_t)4 * 9216 * 2);

    // convert the 4 big MLPs' weights to bf16, transposed (N x K) + N-padding
    int bases[4] = {IT0, IT0 + 10, IT0 + 20, OUTP};
    for (int t = 0; t < 4; ++t) {
        __bf16* W0b = wbf + (size_t)t * 9216;
        __bf16* W1b = W0b + 4096;
        __bf16* Wob = W1b + 4096;
        int nout = (t == 3) ? 1 : 8;
        k_convert_wT<<<(64 * 64 + 255) / 256, 256, 0, stream>>>(F(bases[t] + 0), W0b, 64, 64, 64);
        k_convert_wT<<<(64 * 64 + 255) / 256, 256, 0, stream>>>(F(bases[t] + 1), W1b, 64, 64, 64);
        k_convert_wT<<<(64 * 16 + 255) / 256, 256, 0, stream>>>(F(bases[t] + 8), Wob, 64, nout, 16);
    }

    k_zero<<<1, 32, 0, stream>>>((float*)d_out, (size_t)out_size);

    k_center_embed<<<(NATOM + 255) / 256, 256, 0, stream>>>(species,
        F(EC + 0), F(EC + 2), F(EC + 4), F(EC + 6),
        F(EC + 1), F(EC + 3), F(EC + 5), F(EC + 7),
        F(EC + 8), F(EC + 9), ccen);

    k_pair_setup<<<(NPAIR + 255) / 256, 256, 0, stream>>>(cart, nlist, shifts, nfac, species,
        F(EN + 0), F(EN + 2), F(EN + 4), F(EN + 6),
        F(EN + 1), F(EN + 3), F(EN + 5), F(EN + 7),
        F(EN + 8), F(EN + 9),
        cutd, radial, sphv, wemb, icoef);

    const size_t CORB_N = (size_t)NATOM * 72;
    k_zero<<<(int)((CORB_N + 255) / 256), 256, 0, stream>>>(corbA, CORB_N);
    k_scatter0<<<(NPAIR + 255) / 256, 256, 0, stream>>>(nlist, cutd, radial, sphv, wemb, corbA);
    k_co_density<<<NATOM, 64, 0, stream>>>(corbA, cc_all, 0, ccen, densb);

    const int MT = NATOM / 16;  // 625, exact
    float* corbPrev = corbA;
    float* corbNext = corbB;
    for (int it = 0; it < 3; ++it) {
        const int b = bases[it];
        __bf16* W0b = wbf + (size_t)it * 9216;
        __bf16* W1b = W0b + 4096;
        __bf16* Wob = W1b + 4096;
        k_gemm_bias_wmma<64><<<MT, 32, 0, stream>>>(densb, W0b, F(b + 2), 64, Yf);
        k_ln_silu<<<NATOM, 64, 0, stream>>>(Yf, F(b + 4), F(b + 6), Xb);
        k_gemm_bias_wmma<64><<<MT, 32, 0, stream>>>(Xb, W1b, F(b + 3), 64, Yf);
        k_ln_silu<<<NATOM, 64, 0, stream>>>(Yf, F(b + 5), F(b + 7), Xb);
        k_gemm_bias_wmma<16><<<MT, 32, 0, stream>>>(Xb, Wob, F(b + 9), 8, nnY);

        k_zero<<<(int)((CORB_N + 255) / 256), 256, 0, stream>>>(corbNext, CORB_N);
        k_scatter_iter<<<(NPAIR + 255) / 256, 256, 0, stream>>>(nlist, cutd, radial, sphv,
                                                                nnY, icoef, corbPrev, corbNext);
        k_co_density<<<NATOM, 64, 0, stream>>>(corbNext, cc_all, it + 1, ccen, densb);
        float* tmp = corbPrev; corbPrev = corbNext; corbNext = tmp;
    }

    // output head
    {
        const int b = OUTP;
        __bf16* W0b = wbf + (size_t)3 * 9216;
        __bf16* W1b = W0b + 4096;
        __bf16* Wob = W1b + 4096;
        k_gemm_bias_wmma<64><<<MT, 32, 0, stream>>>(densb, W0b, F(b + 2), 64, Yf);
        k_ln_silu<<<NATOM, 64, 0, stream>>>(Yf, F(b + 4), F(b + 6), Xb);
        k_gemm_bias_wmma<64><<<MT, 32, 0, stream>>>(Xb, W1b, F(b + 3), 64, Yf);
        k_ln_silu<<<NATOM, 64, 0, stream>>>(Yf, F(b + 5), F(b + 7), Xb);
        k_gemm_bias_wmma<16><<<MT, 32, 0, stream>>>(Xb, Wob, F(b + 9), 1, nnY);
    }
    k_final<<<(NATOM + 255) / 256, 256, 0, stream>>>(nnY, cfac, (float*)d_out);
}
